// AttentionBaseline_82523501626025
// MI455X (gfx1250) — matrix operand order repaired and checked by reading, not compiled, not run on hardware
//
#include <hip/hip_runtime.h>
#include <hip/hip_bf16.h>

// ---------------------------------------------------------------------------
// Problem constants (from the reference)
// ---------------------------------------------------------------------------
#define BATCH   64
#define MAXSET  2048
#define EMBD    512
#define HIDDEN  512
#define TARGET  128

typedef __attribute__((ext_vector_type(16))) _Float16 v16h;
typedef __attribute__((ext_vector_type(8)))  _Float16 v8h;
typedef __attribute__((ext_vector_type(8)))  float    v8f;

// ---------------------------------------------------------------------------
// K1a: rowsum[b][s] = sum_e words[b][s][e]   (mask source: valid iff != 0)
// grid (B, S/8), block 256 (8 waves, one row per wave)
// ---------------------------------------------------------------------------
__global__ __launch_bounds__(256) void k_rowsum(const float* __restrict__ words,
                                                float* __restrict__ rowsum) {
    const int b    = blockIdx.x;
    const int s    = blockIdx.y * 8 + (threadIdx.x >> 5);
    const int lane = threadIdx.x & 31;
    const float* row = words + ((size_t)b * MAXSET + s) * EMBD;
    const float4 p0 = *(const float4*)(row + lane * 16 + 0);
    const float4 p1 = *(const float4*)(row + lane * 16 + 4);
    const float4 p2 = *(const float4*)(row + lane * 16 + 8);
    const float4 p3 = *(const float4*)(row + lane * 16 + 12);
    float t = (p0.x + p0.y + p0.z + p0.w) + (p1.x + p1.y + p1.z + p1.w)
            + (p2.x + p2.y + p2.z + p2.w) + (p3.x + p3.y + p3.z + p3.w);
    t += __shfl_xor(t, 1, 32);
    t += __shfl_xor(t, 2, 32);
    t += __shfl_xor(t, 4, 32);
    t += __shfl_xor(t, 8, 32);
    t += __shfl_xor(t, 16, 32);
    if (lane == 0) rowsum[b * MAXSET + s] = t;
}

// ---------------------------------------------------------------------------
// K1b: column partial sums over 64-row chunks (deterministic, no atomics)
// colpart[(b*32 + chunk)*512 + e] = sum_{s in chunk} words[b][s][e]
// grid (B, 32), block 256
// ---------------------------------------------------------------------------
__global__ __launch_bounds__(256) void k_colpart(const float* __restrict__ words,
                                                 float* __restrict__ colpart) {
    const int b = blockIdx.x;
    const int c = blockIdx.y;
    const int tid = threadIdx.x;
    const float* base = words + ((size_t)b * MAXSET + c * 64) * EMBD;
    float r0 = 0.f, r1 = 0.f;
    for (int s = 0; s < 64; ++s) {
        r0 += base[(size_t)s * EMBD + tid];
        r1 += base[(size_t)s * EMBD + tid + 256];
    }
    colpart[((size_t)(b * 32 + c)) * EMBD + tid]       = r0;
    colpart[((size_t)(b * 32 + c)) * EMBD + tid + 256] = r1;
}

// ---------------------------------------------------------------------------
// K2: per-batch length, context, and fused attention bias
//   cb[b][h] = b_att[h] + sum_e context[b][e] * W_att[E+e][h]
// grid B, block 256
// ---------------------------------------------------------------------------
__global__ __launch_bounds__(256) void k_ctx_cb(const float* __restrict__ rowsum,
                                                const float* __restrict__ colpart,
                                                const float* __restrict__ W_att,
                                                const float* __restrict__ b_att,
                                                float* __restrict__ cb) {
    __shared__ float red[256];
    __shared__ float cx[EMBD];
    const int b = blockIdx.x;
    const int tid = threadIdx.x;

    // valid-row count
    float cnt = 0.f;
    for (int s = tid; s < MAXSET; s += 256)
        cnt += (rowsum[b * MAXSET + s] != 0.0f) ? 1.0f : 0.0f;
    red[tid] = cnt;
    __syncthreads();
    for (int off = 128; off > 0; off >>= 1) {
        if (tid < off) red[tid] += red[tid + off];
        __syncthreads();
    }
    const float len = red[0];

    // context = colsum / len
    for (int e = tid; e < EMBD; e += 256) {
        float s = 0.f;
        for (int c = 0; c < 32; ++c)
            s += colpart[((size_t)(b * 32 + c)) * EMBD + e];
        cx[e] = s / len;
    }
    __syncthreads();

    // cb = b_att + context @ W_att[E:, :]
    for (int h = tid; h < HIDDEN; h += 256) {
        float a = b_att[h];
        for (int e = 0; e < EMBD; ++e)
            a += cx[e] * W_att[(size_t)(EMBD + e) * HIDDEN + h];
        cb[b * HIDDEN + h] = a;
    }
}

// ---------------------------------------------------------------------------
// K_cvtB: one-shot convert + swizzle of W_att[:E, :] into f16 slabs.
// Layout: Bh[(ks*512 + n)*32 + kk] = (f16) W_att[ks*32+kk][n]
//   -> per K-step ks, a packed 32 KB slab; each column n is 64 contiguous
//      bytes (its 32 K values), i.e. exactly one WMMA B-fragment pair.
// grid 1024, block 256
// ---------------------------------------------------------------------------
__global__ __launch_bounds__(256) void k_cvtB(const float* __restrict__ W_att,
                                              _Float16* __restrict__ Bh) {
    const int o  = blockIdx.x * 256 + threadIdx.x;   // 0..262143
    const int ks  = o >> 14;                          // / 16384
    const int rem = o & 16383;
    const int n   = rem >> 5;
    const int kk  = rem & 31;
    Bh[o] = (_Float16)W_att[(size_t)(ks * 32 + kk) * HIDDEN + n];
}

// ---------------------------------------------------------------------------
// K3: scores[b][s] = v . tanh( words[b,s,:] @ W_att[:E,:]  + cb[b,:] )
// WMMA f16 16x16x32, f32 accumulate.
//  block = 256 threads = 8 waves: 4 M-tiles x 2 N-halves, 64 rows per block.
//  Each wave: 1 M-tile (16 rows) x 16 N-tiles (256 cols) -> 128 acc VGPRs.
//  B slabs (pre-converted f16, packed 32KB) are streamed into double-buffered
//  LDS via global_load_async_to_lds_b128 (ASYNCcnt), overlapped with compute.
//  LDS layout: column n at byte n*80 (+16B pad per column for bank spread).
// grid B*S/64 = 2048, block 256
// ---------------------------------------------------------------------------
#define BSTRIDE 40   // _Float16 elements per column (32 K + 8 pad = 80 bytes)

__global__ __launch_bounds__(256) void k_scores(const float* __restrict__ words,
                                                const _Float16* __restrict__ Bh,
                                                const float* __restrict__ v,
                                                const float* __restrict__ cb,
                                                float* __restrict__ scores) {
    __shared__ alignas(16) _Float16 ldsB[2][EMBD * BSTRIDE];  // 2 x 40 KB
    __shared__ float spbuf[2][64];

    const int tid   = threadIdx.x;
    const int wave  = tid >> 5;
    const int lane  = tid & 31;
    const int half  = lane >> 4;     // 0: lanes 0-15, 1: lanes 16-31
    const int m     = lane & 15;
    const int mtile = wave >> 1;     // 0..3
    const int nhalf = wave & 1;      // 0..1  (N cols 0..255 / 256..511)

    const int blk    = blockIdx.x;           // 0..2047
    const int b      = blk >> 5;             // / (S/64)
    const int mchunk = blk & 31;
    const int s0     = mchunk * 64;          // block row base in batch b
    const int srow   = s0 + mtile * 16 + m;  // this lane's A row

    const float* arow = words + ((size_t)b * MAXSET + srow) * EMBD;
    const int koffA = half * 8;              // A-matrix lane K offset (16-bit layout)

    // per-thread async-copy addresses: 8 x 16B chunks of the 32KB slab.
    // chunk id = c*256 + tid; packed byte gb -> LDS byte (gb/64)*80 + gb%64
    unsigned goff[8], loff[8];
    #pragma unroll
    for (int c = 0; c < 8; ++c) {
        const unsigned gb = (unsigned)(c * 256 + tid) * 16u;
        goff[c] = gb;
        loff[c] = (gb >> 6) * 80u + (gb & 63u);
    }

    auto issue_slab = [&](int ks, int buf) {
        const unsigned base = (unsigned)(uintptr_t)&ldsB[buf][0];
        const unsigned long long sa =
            (unsigned long long)(uintptr_t)(Bh + (size_t)ks * (EMBD * 32));
        #pragma unroll
        for (int c = 0; c < 8; ++c) {
            const unsigned dst = base + loff[c];
            asm volatile("global_load_async_to_lds_b128 %0, %1, %2"
                         :: "v"(dst), "v"(goff[c]), "s"(sa) : "memory");
        }
    };

    v8f acc[16];
    #pragma unroll
    for (int t = 0; t < 16; ++t)
        #pragma unroll
        for (int i = 0; i < 8; ++i) acc[t][i] = 0.0f;

    // prologue: fetch slab 0
    issue_slab(0, 0);
    asm volatile("s_wait_asynccnt 0x0" ::: "memory");
    __syncthreads();

    for (int ks = 0; ks < EMBD / 32; ++ks) {
        const int buf = ks & 1;
        const int k0  = ks * 32;

        // prefetch next slab into the other buffer while we compute
        if (ks + 1 < EMBD / 32) issue_slab(ks + 1, buf ^ 1);

        // A fragment: 16-bit A 16x32 layout.
        // lanes 0-15:  V0-3 K=0..7,  V4-7 K=16..23
        // lanes 16-31: V0-3 K=8..15, V4-7 K=24..31
        const float4 a0 = *(const float4*)(arow + k0 + koffA + 0);
        const float4 a1 = *(const float4*)(arow + k0 + koffA + 4);
        const float4 a2 = *(const float4*)(arow + k0 + koffA + 16);
        const float4 a3 = *(const float4*)(arow + k0 + koffA + 20);
        v16h A;
        A[0]  = (_Float16)a0.x; A[1]  = (_Float16)a0.y;
        A[2]  = (_Float16)a0.z; A[3]  = (_Float16)a0.w;
        A[4]  = (_Float16)a1.x; A[5]  = (_Float16)a1.y;
        A[6]  = (_Float16)a1.z; A[7]  = (_Float16)a1.w;
        A[8]  = (_Float16)a2.x; A[9]  = (_Float16)a2.y;
        A[10] = (_Float16)a2.z; A[11] = (_Float16)a2.w;
        A[12] = (_Float16)a3.x; A[13] = (_Float16)a3.y;
        A[14] = (_Float16)a3.z; A[15] = (_Float16)a3.w;

        // B fragment per N-tile: 32x16, lanes 0-15 K=0..7,16..23, lanes 16-31 K=8..15,24..31,
        // column N = ntile*16 + m; elem i of v16h = K (i<8 ? half*8+i : 16+half*8+i-8).
        #pragma unroll
        for (int t = 0; t < 16; ++t) {
            const int n = (nhalf * 16 + t) * 16 + m;
            union { v16h v; v8h h[2]; } bu;
            bu.h[0] = *(const v8h*)(&ldsB[buf][n * BSTRIDE + half * 8 + 0]);
            bu.h[1] = *(const v8h*)(&ldsB[buf][n * BSTRIDE + half * 8 + 16]);
            acc[t] = __builtin_amdgcn_wmma_f32_16x16x32_f16(
                false, A, false, bu.v, (short)0, acc[t], false, false);
        }

        // my async loads done, then everyone's (barrier) -> next buffer ready
        asm volatile("s_wait_asynccnt 0x0" ::: "memory");
        __syncthreads();
    }

    // Epilogue: tanh + dot with v, fused per lane.
    // C layout: VGPR i, lanes 0-15 -> row M=i; lanes 16-31 -> row M=8+i.
    float sp[8];
    #pragma unroll
    for (int i = 0; i < 8; ++i) sp[i] = 0.0f;
    #pragma unroll
    for (int t = 0; t < 16; ++t) {
        const int n = (nhalf * 16 + t) * 16 + m;
        const float bias = cb[b * HIDDEN + n];
        const float vn   = v[n];
        #pragma unroll
        for (int i = 0; i < 8; ++i)
            sp[i] += tanhf(acc[t][i] + bias) * vn;
    }
    // reduce across the 16 lanes sharing a row set (xor masks < 16 stay in-half)
    #pragma unroll
    for (int i = 0; i < 8; ++i) {
        sp[i] += __shfl_xor(sp[i], 1, 32);
        sp[i] += __shfl_xor(sp[i], 2, 32);
        sp[i] += __shfl_xor(sp[i], 4, 32);
        sp[i] += __shfl_xor(sp[i], 8, 32);
    }
    if (m == 0) {
        #pragma unroll
        for (int i = 0; i < 8; ++i)
            spbuf[nhalf][mtile * 16 + half * 8 + i] = sp[i];
    }
    __syncthreads();
    if (tid < 64)
        scores[(size_t)b * MAXSET + s0 + tid] = spbuf[0][tid] + spbuf[1][tid];
}

// ---------------------------------------------------------------------------
// K4: masked softmax over S per batch, in place on scores.
// grid B, block 256
// ---------------------------------------------------------------------------
__global__ __launch_bounds__(256) void k_softmax(const float* __restrict__ rowsum,
                                                 float* __restrict__ scores) {
    __shared__ float ls[MAXSET];
    __shared__ float red[256];
    const int b = blockIdx.x;
    const int tid = threadIdx.x;
    for (int s = tid; s < MAXSET; s += 256) {
        const float val = scores[b * MAXSET + s];
        ls[s] = (rowsum[b * MAXSET + s] != 0.0f) ? val : -1e30f;
    }
    __syncthreads();
    float mx = -1e30f;
    for (int s = tid; s < MAXSET; s += 256) mx = fmaxf(mx, ls[s]);
    red[tid] = mx;
    __syncthreads();
    for (int off = 128; off > 0; off >>= 1) {
        if (tid < off) red[tid] = fmaxf(red[tid], red[tid + off]);
        __syncthreads();
    }
    mx = red[0];
    __syncthreads();
    float es = 0.f;
    for (int s = tid; s < MAXSET; s += 256) es += __expf(ls[s] - mx);
    red[tid] = es;
    __syncthreads();
    for (int off = 128; off > 0; off >>= 1) {
        if (tid < off) red[tid] += red[tid + off];
        __syncthreads();
    }
    const float inv = 1.0f / red[0];
    for (int s = tid; s < MAXSET; s += 256)
        scores[b * MAXSET + s] = __expf(ls[s] - mx) * inv;
}

// ---------------------------------------------------------------------------
// K5a: rep partials over 256-row chunks: repp[(b*8+c)*512 + e]
// grid (B, 8), block 256
// ---------------------------------------------------------------------------
__global__ __launch_bounds__(256) void k_repp(const float* __restrict__ words,
                                              const float* __restrict__ attn,
                                              float* __restrict__ repp) {
    const int b = blockIdx.x;
    const int c = blockIdx.y;
    const int tid = threadIdx.x;
    float r0 = 0.f, r1 = 0.f;
    const float* base = words + ((size_t)b * MAXSET + c * 256) * EMBD;
    const float* av   = attn + b * MAXSET + c * 256;
    for (int s = 0; s < 256; ++s) {
        const float a = av[s];
        r0 += a * base[(size_t)s * EMBD + tid];
        r1 += a * base[(size_t)s * EMBD + tid + 256];
    }
    repp[((size_t)(b * 8 + c)) * EMBD + tid]       = r0;
    repp[((size_t)(b * 8 + c)) * EMBD + tid + 256] = r1;
}

// K5b: reduce the 8 partials -> rep[B][E]
__global__ __launch_bounds__(256) void k_repred(const float* __restrict__ repp,
                                                float* __restrict__ rep) {
    const int idx = blockIdx.x * 256 + threadIdx.x;   // 0..32767
    const int b = idx >> 9;
    const int e = idx & 511;
    float s = 0.f;
    for (int c = 0; c < 8; ++c)
        s += repp[((size_t)(b * 8 + c)) * EMBD + e];
    rep[idx] = s;
}

// ---------------------------------------------------------------------------
// K6: small MLP layers, one output per thread (coalesced over Out).
// ---------------------------------------------------------------------------
__global__ __launch_bounds__(256) void k_mlp(const float* __restrict__ X,
                                             const float* __restrict__ W,
                                             const float* __restrict__ bias,
                                             float* __restrict__ Y,
                                             int In, int Out, int do_relu) {
    const int idx = blockIdx.x * 256 + threadIdx.x;
    const int b = idx / Out;
    const int o = idx % Out;
    float a = bias[o];
    const float* x = X + (size_t)b * In;
    for (int e = 0; e < In; ++e)
        a += x[e] * W[(size_t)e * Out + o];
    Y[idx] = do_relu ? fmaxf(a, 0.0f) : a;
}

// ---------------------------------------------------------------------------
// Launcher
// ---------------------------------------------------------------------------
extern "C" void kernel_launch(void* const* d_in, const int* in_sizes, int n_in,
                              void* d_out, int out_size, void* d_ws, size_t ws_size,
                              hipStream_t stream) {
    const float* words = (const float*)d_in[0];
    const float* W_att = (const float*)d_in[1];
    const float* b_att = (const float*)d_in[2];
    const float* v     = (const float*)d_in[3];
    const float* W1    = (const float*)d_in[4];
    const float* b1    = (const float*)d_in[5];
    const float* W2    = (const float*)d_in[6];
    const float* b2    = (const float*)d_in[7];
    const float* W3    = (const float*)d_in[8];
    const float* b3    = (const float*)d_in[9];

    float* ws = (float*)d_ws;
    float* rowsum  = ws;                            // B*S       = 131072
    float* colpart = rowsum + BATCH * MAXSET;       // B*32*E    = 1048576
    float* cbuf    = colpart + BATCH * 32 * EMBD;   // B*H       = 32768
    float* scores  = cbuf + BATCH * HIDDEN;         // B*S       = 131072 (attn in place)
    float* repp    = scores + BATCH * MAXSET;       // B*8*E     = 262144
    float* rep     = repp + BATCH * 8 * EMBD;       // B*E       = 32768
    float* h1      = rep + BATCH * EMBD;            // B*H       = 32768
    float* h2      = h1 + BATCH * HIDDEN;           // B*H       = 32768
    _Float16* Bh   = (_Float16*)(h2 + BATCH * HIDDEN); // 16*512*32 f16 = 512 KB

    k_rowsum  <<<dim3(BATCH, MAXSET / 8), 256, 0, stream>>>(words, rowsum);
    k_colpart <<<dim3(BATCH, 32),         256, 0, stream>>>(words, colpart);
    k_ctx_cb  <<<BATCH,                   256, 0, stream>>>(rowsum, colpart, W_att, b_att, cbuf);
    k_cvtB    <<<(16 * EMBD * 32) / 256,  256, 0, stream>>>(W_att, Bh);
    k_scores  <<<(BATCH * MAXSET) / 64,   256, 0, stream>>>(words, Bh, v, cbuf, scores);
    k_softmax <<<BATCH,                   256, 0, stream>>>(rowsum, scores);
    k_repp    <<<dim3(BATCH, 8),          256, 0, stream>>>(words, scores, repp);
    k_repred  <<<(BATCH * EMBD) / 256,    256, 0, stream>>>(repp, rep);
    k_mlp     <<<(BATCH * HIDDEN) / 256,  256, 0, stream>>>(rep, W1, b1, h1, EMBD,   HIDDEN, 1);
    k_mlp     <<<(BATCH * HIDDEN) / 256,  256, 0, stream>>>(h1,  W2, b2, h2, HIDDEN, HIDDEN, 1);
    k_mlp     <<<(BATCH * TARGET) / 256,  256, 0, stream>>>(h2,  W3, b3, (float*)d_out, HIDDEN, TARGET, 0);
}
